// MoE_layer_torch_26044681683726
// MI455X (gfx1250) — matrix operand order repaired and checked by reading, not compiled, not run on hardware
//
#include <hip/hip_runtime.h>
#include <hip/hip_bf16.h>

// ---- problem constants (match reference) ----
#define NTOK 2048
#define HID  1024
#define FFN  2048
#define NEXP 8
#define TOPK 2
#define NROWS (NTOK*TOPK)          // 4096 dispatched rows
#define MT   128                   // M tile (rows)
#define NT   128                   // N tile (cols)
#define KT   32                    // K step (matches wmma 16x16x32 bf16)
#define NROWS_PAD 5120             // 4096 + 8*127 rounded to 128
#define MAXMT (NROWS_PAD/MT)       // 40 m-tiles
#define TEXP_CAP 64
#define KPAD (KT+8)                // LDS row stride in ushorts (80B, 16B aligned)

typedef __attribute__((ext_vector_type(16))) __bf16        v16bf;
typedef __attribute__((ext_vector_type(2)))  __bf16        v2bf;
typedef __attribute__((ext_vector_type(8)))  float         v8f;
typedef __attribute__((ext_vector_type(8)))  unsigned short us8;

union Frag { us8 h[2]; v16bf v; };

// single f32 -> bf16 (RNE fptrunc)
__device__ inline unsigned short f2bf(float f) {
  __bf16 h = (__bf16)f;
  return __builtin_bit_cast(unsigned short, h);
}
// pack two f32 -> one dword of two bf16; native 2-vector build lowers
// to a single v_cvt_pk_bf16_f32 (RNE, same numerics as scalar path)
__device__ inline unsigned int pk2bf(float lo, float hi) {
  v2bf p;
  p.x = (__bf16)lo;
  p.y = (__bf16)hi;
  return __builtin_bit_cast(unsigned int, p);
}
__device__ inline float gelu_exact(float x) {
  return 0.5f * x * (1.0f + erff(x * 0.70710678118654752f));
}

// ------------------------------------------------------------------
// Kernel 0: routing. Histogram expert ids, pad each expert segment to
// a multiple of MT, scatter dispatched-row ids, emit tile->expert map.
// ------------------------------------------------------------------
__global__ __launch_bounds__(256)
void moe_route(const int* __restrict__ topk_ids,
               int* __restrict__ sorted_rows,
               int* __restrict__ tile_expert)
{
  __shared__ int cnt[NEXP];
  __shared__ int cur[NEXP];
  const int tid = threadIdx.x;
  if (tid < NEXP) cnt[tid] = 0;
  __syncthreads();
  for (int r = tid; r < NROWS; r += 256)
    atomicAdd(&cnt[topk_ids[r]], 1);
  __syncthreads();
  if (tid == 0) {
    int off = 0, tiles = 0;
    for (int e = 0; e < NEXP; ++e) {
      cur[e] = off;
      int t = (cnt[e] + MT - 1) / MT;
      for (int i = 0; i < t; ++i) tile_expert[tiles + i] = e;
      tiles += t;
      off   += t * MT;
    }
    for (int t = tiles; t < TEXP_CAP; ++t) tile_expert[t] = -1;
  }
  __syncthreads();
  for (int i = tid; i < NROWS_PAD; i += 256) sorted_rows[i] = -1;
  __threadfence_block();
  __syncthreads();
  for (int r = tid; r < NROWS; r += 256) {
    int e = topk_ids[r];
    int p = atomicAdd(&cur[e], 1);
    sorted_rows[p] = r;
  }
}

// ------------------------------------------------------------------
// Kernel 1: zero the output (harness poisons it; atomics accumulate).
// ------------------------------------------------------------------
__global__ __launch_bounds__(256)
void moe_zero(float* __restrict__ out, int n)
{
  int i = blockIdx.x * 256 + threadIdx.x;
  if (i < n) out[i] = 0.0f;
}

// ------------------------------------------------------------------
// Kernel 2: inter = gelu(gather(x) @ W0[e])  -> bf16, sorted row order
// 128x128 block tile; 8 waves as 4(m) x 2(n); each wave 32x64 = 2x4
// v_wmma_f32_16x16x32_bf16 accumulators. Register-prefetch + LDS
// double-buffer: one barrier per K-step. Padding rows (-1) gather row
// 0 unmasked — their results only land in rows GEMM2's epilogue drops.
// B staging: each thread owns one n-column; 16 row-pitch-strided dword
// loads (coalesced across the wave), 8 pairwise cvt_pk_bf16, two
// ds_store_b128 into the [n][k] LDS layout.
// ------------------------------------------------------------------
__global__ __launch_bounds__(256)
void moe_gemm1(const float* __restrict__ x,
               const float* __restrict__ w0,
               const int*   __restrict__ sorted_rows,
               const int*   __restrict__ tile_expert,
               unsigned short* __restrict__ inter)
{
  const int mt = blockIdx.y;
  const int e  = tile_expert[mt];
  if (e < 0) return;
  const int n0   = blockIdx.x * NT;
  const int tid  = threadIdx.x;
  const int wave = tid >> 5, lane = tid & 31;
  const int wm = wave >> 1, wn = wave & 1;
  const int half = lane >> 4, l16 = lane & 15;

  __shared__ unsigned short As[2][MT][KPAD];
  __shared__ unsigned short Bs[2][NT][KPAD];

  const float* w0e = w0 + (size_t)e * HID * FFN;

  // ---- k-invariant staging descriptors ----
  // A: 4 slots x float4 over 128 rows x 32 k (padding rows -> row 0)
  const float* aptr[4];  int ar[4], akc[4];
#pragma unroll
  for (int s = 0; s < 4; ++s) {
    int slot = tid + s * 256;
    int r  = slot >> 3;
    int kc = (slot & 7) << 2;
    int id = sorted_rows[mt * MT + r];
    if (id < 0) id = 0;
    aptr[s] = x + (size_t)id * HID + kc;
    ar[s] = r; akc[s] = kc;
  }
  // B: one n-column per thread, 16 k's (two threads per column)
  const int bn  = tid & 127;
  const int bkq = (tid >> 7) << 4;           // 0 or 16
  const float* bcol = w0e + (size_t)bkq * FFN + n0 + bn;

  v8f acc[2][4];
#pragma unroll
  for (int mi = 0; mi < 2; ++mi)
#pragma unroll
    for (int ni = 0; ni < 4; ++ni)
#pragma unroll
      for (int r = 0; r < 8; ++r) acc[mi][ni][r] = 0.0f;

  // ---- prologue: load first tiles into registers ----
  float4 areg[4];
  float  bv[16];
#pragma unroll
  for (int s = 0; s < 4; ++s) { areg[s] = *(const float4*)aptr[s]; aptr[s] += KT; }
#pragma unroll
  for (int i = 0; i < 16; ++i) bv[i] = bcol[(size_t)i * FFN];
  bcol += (size_t)KT * FFN;

  const int nk = HID / KT;
  for (int ks = 0; ks < nk; ++ks) {
    const int buf = ks & 1;
    // convert + store current registers into LDS buffer `buf`
#pragma unroll
    for (int s = 0; s < 4; ++s) {
      uint2 p;
      p.x = pk2bf(areg[s].x, areg[s].y);
      p.y = pk2bf(areg[s].z, areg[s].w);
      *(uint2*)&As[buf][ar[s]][akc[s]] = p;
    }
    {
      uint4 q0, q1;
      q0.x = pk2bf(bv[0],  bv[1]);  q0.y = pk2bf(bv[2],  bv[3]);
      q0.z = pk2bf(bv[4],  bv[5]);  q0.w = pk2bf(bv[6],  bv[7]);
      q1.x = pk2bf(bv[8],  bv[9]);  q1.y = pk2bf(bv[10], bv[11]);
      q1.z = pk2bf(bv[12], bv[13]); q1.w = pk2bf(bv[14], bv[15]);
      *(uint4*)&Bs[buf][bn][bkq]     = q0;
      *(uint4*)&Bs[buf][bn][bkq + 8] = q1;
    }
    // prefetch next tiles (overlaps with this step's WMMAs)
    if (ks + 1 < nk) {
#pragma unroll
      for (int s = 0; s < 4; ++s) { areg[s] = *(const float4*)aptr[s]; aptr[s] += KT; }
#pragma unroll
      for (int i = 0; i < 16; ++i) bv[i] = bcol[(size_t)i * FFN];
      bcol += (size_t)KT * FFN;
    }
    __syncthreads();
    // compute from LDS buffer `buf`
    Frag a[2], b[4];
#pragma unroll
    for (int mi = 0; mi < 2; ++mi) {
      int row = wm * 32 + mi * 16 + l16;
      a[mi].h[0] = *(const us8*)&As[buf][row][half * 16];
      a[mi].h[1] = *(const us8*)&As[buf][row][half * 16 + 8];
    }
#pragma unroll
    for (int ni = 0; ni < 4; ++ni) {
      int col = wn * 64 + ni * 16 + l16;
      b[ni].h[0] = *(const us8*)&Bs[buf][col][half * 16];
      b[ni].h[1] = *(const us8*)&Bs[buf][col][half * 16 + 8];
    }
#pragma unroll
    for (int mi = 0; mi < 2; ++mi)
#pragma unroll
      for (int ni = 0; ni < 4; ++ni)
        acc[mi][ni] = __builtin_amdgcn_wmma_f32_16x16x32_bf16(
            false, a[mi].v, false, b[ni].v, (short)0, acc[mi][ni], false, false);
  }

  // epilogue: exact GeLU, store bf16 intermediate in sorted-row order
#pragma unroll
  for (int mi = 0; mi < 2; ++mi)
#pragma unroll
    for (int ni = 0; ni < 4; ++ni)
#pragma unroll
      for (int r = 0; r < 8; ++r) {
        int m    = r + half * 8;
        int grow = mt * MT + wm * 32 + mi * 16 + m;
        int gcol = n0 + wn * 64 + ni * 16 + l16;
        inter[(size_t)grow * FFN + gcol] = f2bf(gelu_exact(acc[mi][ni][r]));
      }
}

// ------------------------------------------------------------------
// Kernel 3: out[token] += inter @ W1[e]   (scatter-add combine, K=2)
// Same pipelined structure; A is already bf16 in sorted order.
// ------------------------------------------------------------------
__global__ __launch_bounds__(256)
void moe_gemm2(const unsigned short* __restrict__ inter,
               const float* __restrict__ w1,
               const int*   __restrict__ sorted_rows,
               const int*   __restrict__ tile_expert,
               float* __restrict__ out)
{
  const int mt = blockIdx.y;
  const int e  = tile_expert[mt];
  if (e < 0) return;
  const int n0   = blockIdx.x * NT;         // over HID
  const int tid  = threadIdx.x;
  const int wave = tid >> 5, lane = tid & 31;
  const int wm = wave >> 1, wn = wave & 1;
  const int half = lane >> 4, l16 = lane & 15;

  __shared__ unsigned short As[2][MT][KPAD];
  __shared__ unsigned short Bs[2][NT][KPAD];
  __shared__ int rowids[MT];

  for (int i = tid; i < MT; i += 256) rowids[i] = sorted_rows[mt * MT + i];

  const float* w1e = w1 + (size_t)e * FFN * HID;

  // ---- k-invariant staging descriptors ----
  const unsigned short* aptr[2];  int ar[2], ac[2];
#pragma unroll
  for (int s = 0; s < 2; ++s) {
    int slot = tid + s * 256;                // 512 us8 chunks: 128 rows x 4
    int r = slot >> 2;
    int c = (slot & 3) << 3;
    aptr[s] = inter + (size_t)(mt * MT + r) * FFN + c;
    ar[s] = r; ac[s] = c;
  }
  const int bn  = tid & 127;
  const int bkq = (tid >> 7) << 4;
  const float* bcol = w1e + (size_t)bkq * HID + n0 + bn;

  v8f acc[2][4];
#pragma unroll
  for (int mi = 0; mi < 2; ++mi)
#pragma unroll
    for (int ni = 0; ni < 4; ++ni)
#pragma unroll
      for (int r = 0; r < 8; ++r) acc[mi][ni][r] = 0.0f;

  // ---- prologue ----
  us8   areg[2];
  float bv[16];
#pragma unroll
  for (int s = 0; s < 2; ++s) { areg[s] = *(const us8*)aptr[s]; aptr[s] += KT; }
#pragma unroll
  for (int i = 0; i < 16; ++i) bv[i] = bcol[(size_t)i * HID];
  bcol += (size_t)KT * HID;

  const int nk = FFN / KT;
  for (int ks = 0; ks < nk; ++ks) {
    const int buf = ks & 1;
#pragma unroll
    for (int s = 0; s < 2; ++s)
      *(us8*)&As[buf][ar[s]][ac[s]] = areg[s];
    {
      uint4 q0, q1;
      q0.x = pk2bf(bv[0],  bv[1]);  q0.y = pk2bf(bv[2],  bv[3]);
      q0.z = pk2bf(bv[4],  bv[5]);  q0.w = pk2bf(bv[6],  bv[7]);
      q1.x = pk2bf(bv[8],  bv[9]);  q1.y = pk2bf(bv[10], bv[11]);
      q1.z = pk2bf(bv[12], bv[13]); q1.w = pk2bf(bv[14], bv[15]);
      *(uint4*)&Bs[buf][bn][bkq]     = q0;
      *(uint4*)&Bs[buf][bn][bkq + 8] = q1;
    }
    if (ks + 1 < nk) {
#pragma unroll
      for (int s = 0; s < 2; ++s) { areg[s] = *(const us8*)aptr[s]; aptr[s] += KT; }
#pragma unroll
      for (int i = 0; i < 16; ++i) bv[i] = bcol[(size_t)i * HID];
      bcol += (size_t)KT * HID;
    }
    __syncthreads();
    Frag a[2], b[4];
#pragma unroll
    for (int mi = 0; mi < 2; ++mi) {
      int row = wm * 32 + mi * 16 + l16;
      a[mi].h[0] = *(const us8*)&As[buf][row][half * 16];
      a[mi].h[1] = *(const us8*)&As[buf][row][half * 16 + 8];
    }
#pragma unroll
    for (int ni = 0; ni < 4; ++ni) {
      int col = wn * 64 + ni * 16 + l16;
      b[ni].h[0] = *(const us8*)&Bs[buf][col][half * 16];
      b[ni].h[1] = *(const us8*)&Bs[buf][col][half * 16 + 8];
    }
#pragma unroll
    for (int mi = 0; mi < 2; ++mi)
#pragma unroll
      for (int ni = 0; ni < 4; ++ni)
        acc[mi][ni] = __builtin_amdgcn_wmma_f32_16x16x32_bf16(
            false, a[mi].v, false, b[ni].v, (short)0, acc[mi][ni], false, false);
  }

  // epilogue: scatter-add into out[token] (order-independent for K=2)
#pragma unroll
  for (int mi = 0; mi < 2; ++mi)
#pragma unroll
    for (int ni = 0; ni < 4; ++ni)
#pragma unroll
      for (int r = 0; r < 8; ++r) {
        int m  = r + half * 8;
        int lr = wm * 32 + mi * 16 + m;
        int id = rowids[lr];
        if (id >= 0) {
          int token = id / TOPK;
          int gcol  = n0 + wn * 64 + ni * 16 + l16;
          unsafeAtomicAdd(&out[(size_t)token * HID + gcol], acc[mi][ni][r]);
        }
      }
}

// ------------------------------------------------------------------
extern "C" void kernel_launch(void* const* d_in, const int* in_sizes, int n_in,
                              void* d_out, int out_size, void* d_ws, size_t ws_size,
                              hipStream_t stream) {
  const float* x    = (const float*)d_in[0];   // [N,H]
  const float* w0   = (const float*)d_in[1];   // [E,H,F]
  const float* w1   = (const float*)d_in[2];   // [E,F,H]
  const int*   topk = (const int*)d_in[3];     // [N,K]
  float* out = (float*)d_out;                  // [N,H]

  int* sorted_rows = (int*)d_ws;                              // 5120 ints
  int* tile_expert = sorted_rows + NROWS_PAD;                 // 64 ints
  unsigned short* inter =
      (unsigned short*)(tile_expert + TEXP_CAP);              // 5120*2048 bf16, 16B aligned

  moe_route<<<1, 256, 0, stream>>>(topk, sorted_rows, tile_expert);
  moe_zero<<<(NTOK * HID + 255) / 256, 256, 0, stream>>>(out, NTOK * HID);
  moe_gemm1<<<dim3(FFN / NT, MAXMT), 256, 0, stream>>>(x, w0, sorted_rows,
                                                       tile_expert, inter);
  moe_gemm2<<<dim3(HID / NT, MAXMT), 256, 0, stream>>>(inter, w1, sorted_rows,
                                                       tile_expert, out);
}
